// CrossAttention_31447750541958
// MI455X (gfx1250) — matrix operand order repaired and checked by reading, not compiled, not run on hardware
//
#include <hip/hip_runtime.h>
#include <hip/hip_bf16.h>

// ---------------------------------------------------------------------------
// CDNA5 (gfx1250) cross-attention: bf16 WMMA GEMMs + flash attention with
// TDM (tensor_load_to_lds) double-buffered K feed. wave32.
// ---------------------------------------------------------------------------

typedef __attribute__((ext_vector_type(16))) __bf16 bf16x16;
typedef __attribute__((ext_vector_type(8)))  __bf16 bf16x8;
typedef __attribute__((ext_vector_type(8)))  float  f32x8;

#if defined(__AMDGCN__) && __has_builtin(__builtin_amdgcn_tensor_load_to_lds) && \
    __has_builtin(__builtin_amdgcn_s_wait_tensorcnt)
#define USE_TDM 1
typedef unsigned uint32x4 __attribute__((ext_vector_type(4)));
typedef int      int32x8  __attribute__((ext_vector_type(8)));
typedef int      int32x4  __attribute__((ext_vector_type(4)));
#else
#define USE_TDM 0
#endif

// --- float -> bf16 round-to-nearest-even
__device__ __forceinline__ __bf16 f2bf(float f) {
    union { float f; unsigned u; } in;
    in.f = f;
    unsigned u = in.u + 0x7FFFu + ((in.u >> 16) & 1u);
    union { unsigned short s; __bf16 b; } out;
    out.s = (unsigned short)(u >> 16);
    return out.b;
}

__device__ __forceinline__ __bf16 cvt_in(float v)  { return f2bf(v); }
__device__ __forceinline__ __bf16 cvt_in(__bf16 v) { return v; }

__device__ __forceinline__ void store_out(__bf16* p, float v) { *p = f2bf(v); }
__device__ __forceinline__ void store_out(float*  p, float v) { *p = v; }

// --- A fragment (16x32 bf16, M x K). ISA layout:
//   lanes 0-15 : row M=lane,    elems 0-7 -> K 0-7,   elems 8-15 -> K 16-23
//   lanes16-31 : row M=lane-16, elems 0-7 -> K 8-15,  elems 8-15 -> K 24-31
__device__ __forceinline__ bf16x16
load_frag_a(const __bf16* base, int stride, int m0, int k0, int lane) {
    const int row   = m0 + (lane & 15);
    const int khalf = lane >> 4;
    const __bf16* p = base + row * stride + k0 + 8 * khalf;
    union { bf16x16 v; bf16x8 h[2]; } u;
    u.h[0] = *(const bf16x8*)(p);        // K = k0+8h .. +7
    u.h[1] = *(const bf16x8*)(p + 16);   // K = k0+16+8h .. +7
    return u.v;
}

// --- B fragment (32x16 bf16, K x N), sourced from LDS stored as Bt[N][K]:
//   lanes 0-15 : col N=lane, elems 0-7 -> K 0-7, 8-15 -> K 16-23 ; lanes 16-31 -> K 8-15, 24-31
__device__ __forceinline__ bf16x16
load_frag_b(const __bf16* baseT, int stride, int n0, int k0, int lane) {
    const int n     = n0 + (lane & 15);
    const int khalf = lane >> 4;
    const __bf16* p = baseT + n * stride + k0 + 8 * khalf;
    union { bf16x16 v; bf16x8 h[2]; } u;
    u.h[0] = *(const bf16x8*)(p);
    u.h[1] = *(const bf16x8*)(p + 16);
    return u.v;
}

__device__ __forceinline__ f32x8 wmma_bf16(bf16x16 a, bf16x16 b, f32x8 c) {
    return __builtin_amdgcn_wmma_f32_16x16x32_bf16(false, a, false, b,
                                                   (short)0, c, false, false);
}

// ===========================================================================
// Kernel 1/3: Out[M x 512] = A[M x 512] * W[512 x 512] + bias
// ===========================================================================
template <typename TIn, typename TOut>
__global__ void __launch_bounds__(256)
gemm_bias_wmma(const TIn* __restrict__ A, const float* __restrict__ W,
               const float* __restrict__ bias, TOut* __restrict__ Out) {
    constexpr int N = 512, KD = 512;
    constexpr int BM = 128, BN = 128, BK = 32, LDT = BK + 8;

    __shared__ __bf16 As[BM * LDT];   // [m][k]
    __shared__ __bf16 Bt[BN * LDT];   // [n][k]  (W tile transposed)

    const int tid  = threadIdx.x;
    const int lane = tid & 31;
    const int wave = tid >> 5;
    const int wm   = wave >> 1;
    const int wn   = wave & 1;
    const int row0 = blockIdx.y * BM;
    const int col0 = blockIdx.x * BN;

    f32x8 acc[2][4];
    const f32x8 z = {0.f, 0.f, 0.f, 0.f, 0.f, 0.f, 0.f, 0.f};
#pragma unroll
    for (int i = 0; i < 2; i++)
#pragma unroll
        for (int j = 0; j < 4; j++) acc[i][j] = z;

    for (int k0 = 0; k0 < KD; k0 += BK) {
        for (int i = tid; i < BM * BK; i += 256) {
            int m = i >> 5, k = i & (BK - 1);
            As[m * LDT + k] = cvt_in(A[(size_t)(row0 + m) * KD + k0 + k]);
        }
        for (int i = tid; i < BN * BK; i += 256) {
            int k = i >> 7, n = i & (BN - 1);
            Bt[n * LDT + k] = f2bf(W[(size_t)(k0 + k) * N + col0 + n]);
        }
        __syncthreads();

        bf16x16 a0 = load_frag_a(As, LDT, wm * 32,      0, lane);
        bf16x16 a1 = load_frag_a(As, LDT, wm * 32 + 16, 0, lane);
#pragma unroll
        for (int j = 0; j < 4; j++) {
            bf16x16 b = load_frag_b(Bt, LDT, wn * 64 + j * 16, 0, lane);
            acc[0][j] = wmma_bf16(a0, b, acc[0][j]);
            acc[1][j] = wmma_bf16(a1, b, acc[1][j]);
        }
        __syncthreads();
    }

    const int khalf = lane >> 4, colL = lane & 15;
#pragma unroll
    for (int i = 0; i < 2; i++) {
#pragma unroll
        for (int j = 0; j < 4; j++) {
            const int col = col0 + wn * 64 + j * 16 + colL;
            const float bv = bias[col];
#pragma unroll
            for (int r = 0; r < 8; r++) {
                const int row = row0 + wm * 32 + i * 16 + r + 8 * khalf;
                store_out(&Out[(size_t)row * N + col], acc[i][j][r] + bv);
            }
        }
    }
}

// ===========================================================================
// Kernel 2: flash attention.
// One workgroup = (batch, 64 query rows); Q resident in LDS; loop S in 64-row
// blocks. K blocks double-buffered and fed by the Tensor Data Mover.
// ===========================================================================
constexpr int FA_D   = 512;
constexpr int FA_BQ  = 64;
constexpr int FA_BS  = 64;
constexpr int QS_STR = FA_D + 8;   // 520 bf16 -> 1040B rows (16B aligned)
constexpr int VT_STR = FA_BS + 8;  // 72 bf16 -> 144B rows
constexpr int SS_STR = FA_BS + 4;  // 68 f32  -> 272B rows
constexpr int PS_STR = FA_BS + 8;  // 72 bf16
constexpr size_t FA_SMEM =
    (size_t)FA_BQ * QS_STR * 2 +      // Qs
    (size_t)FA_BS * QS_STR * 2 * 2 +  // Ks[2] (double buffer for TDM overlap)
    (size_t)FA_D  * VT_STR * 2 +      // Vt
    (size_t)FA_BQ * SS_STR * 4 +      // Ss
    (size_t)FA_BQ * PS_STR * 2 +      // Ps
    (size_t)FA_BQ * 4 * 3 +           // m, l, alpha
    (size_t)FA_BQ * 4 * 4;            // reduction scratch
// = 301,824 bytes (< 320KB LDS per workgroup on CDNA5)

#if USE_TDM
// TDM 2-D tile load: tile = FA_D x FA_BS bf16 rows, LDS rows padded to QS_STR
// via the D# pad feature (pad 4 DWORDs every 256 DWORDs = 16B per 1024B row).
// D# layout per CDNA5 ISA.
// NOTE: this toolchain exposes the 6-arg builtin:
//   (uint32x4 g0, int32x8 g1, int32x4 g2, int32x4 g3, int32x8 extra, int cpol)
__device__ __forceinline__ void tdm_load_k_tile(unsigned lds_addr, const __bf16* gsrc) {
    const unsigned long long ga = (unsigned long long)(uintptr_t)gsrc;
    uint32x4 g0;
    g0[0] = 1u;                                        // count=1 (valid user D#)
    g0[1] = lds_addr;                                  // lds_addr (bytes)
    g0[2] = (unsigned)(ga & 0xFFFFFFFFull);            // global_addr[31:0]
    g0[3] = (unsigned)((ga >> 32) & 0x01FFFFFFull)     // global_addr[56:32]
          | 0x80000000u;                               // type=2 ("image")
    const unsigned tensor_d0 = FA_D;                   // elements per row
    const unsigned tensor_d1 = 1u << 20;               // plenty of rows (no OOB)
    const unsigned tile_d0   = FA_D;
    const unsigned tile_d1   = FA_BS;
    const unsigned stride0   = FA_D;                   // row stride, elements
    int32x8 g1;
    g1[0] = (int)((1u << 16)        // data_size = 1 -> 2 bytes (bf16)
                | (1u << 20)        // pad_enable
                | (7u << 22)        // pad_interval: every 256 DWORDs
                | (3u << 25));      // pad_amount: 4 DWORDs
    g1[1] = (int)((tensor_d0 & 0xFFFFu) << 16);                       // dim0[15:0]
    g1[2] = (int)(((tensor_d0 >> 16) & 0xFFFFu)
                | ((tensor_d1 & 0xFFFFu) << 16));                     // dim0[31:16]|dim1[15:0]
    g1[3] = (int)(((tensor_d1 >> 16) & 0xFFFFu) | (tile_d0 << 16));   // dim1[31:16]|tile0
    g1[4] = (int)(tile_d1);                                           // tile1 | tile2=0
    g1[5] = (int)(stride0);                                           // dim0_stride[31:0]
    g1[6] = 0;                                                        // stride0 hi|stride1 lo
    g1[7] = 0;
    const int32x4 gz4 = {0, 0, 0, 0};                  // groups 2/3 unused (2D)
    const int32x8 gz8 = {0, 0, 0, 0, 0, 0, 0, 0};
    __builtin_amdgcn_tensor_load_to_lds(g0, g1, gz4, gz4, gz8, 0);
}
#endif

__device__ __forceinline__ void load_k_block_manual(__bf16* dst, const __bf16* src, int tid) {
    for (int i = tid; i < FA_BS * (FA_D / 8); i += 256) {
        int r = i >> 6, c = i & 63;
        *(bf16x8*)(dst + r * QS_STR + c * 8) =
            *(const bf16x8*)(src + (size_t)r * FA_D + c * 8);
    }
}

__global__ void __launch_bounds__(256)
flash_attn_wmma(const __bf16* __restrict__ Q, const __bf16* __restrict__ K,
                const __bf16* __restrict__ V, __bf16* __restrict__ Ctx) {
    extern __shared__ char smem[];
    __bf16* Qs   = (__bf16*)smem;
    __bf16* Ks0  = Qs  + FA_BQ * QS_STR;
    __bf16* Ks1  = Ks0 + FA_BS * QS_STR;
    __bf16* Vt   = Ks1 + FA_BS * QS_STR;          // [d][s]
    float*  Ss   = (float*)(Vt + FA_D * VT_STR);  // [q][s]
    __bf16* Ps   = (__bf16*)(Ss + FA_BQ * SS_STR);
    float*  mrow = (float*)(Ps + FA_BQ * PS_STR);
    float*  lrow = mrow + FA_BQ;
    float*  arow = lrow + FA_BQ;
    float*  red  = arow + FA_BQ;                  // [64][4]

    const int b    = blockIdx.y;
    const int q0   = blockIdx.x * FA_BQ;
    const size_t boff = (size_t)b * 4096 * FA_D;
    const int tid  = threadIdx.x;
    const int lane = tid & 31;
    const int wave = tid >> 5;       // 8 waves; wave owns d cols [wave*64, +64)

#if USE_TDM
    if (wave == 0)  // kick off DMA of K block 0 before anything else
        tdm_load_k_tile((unsigned)(uintptr_t)Ks0, K + boff);
#endif

    // Load resident Q block (16B chunks).
    for (int i = tid; i < FA_BQ * (FA_D / 8); i += 256) {
        int r = i >> 6, c = i & 63;
        *(bf16x8*)(Qs + r * QS_STR + c * 8) =
            *(const bf16x8*)(Q + boff + (size_t)(q0 + r) * FA_D + c * 8);
    }
    if (tid < FA_BQ) { mrow[tid] = -1e30f; lrow[tid] = 0.f; }

    f32x8 oacc[4][4];
    const f32x8 z = {0.f, 0.f, 0.f, 0.f, 0.f, 0.f, 0.f, 0.f};
#pragma unroll
    for (int i = 0; i < 4; i++)
#pragma unroll
        for (int j = 0; j < 4; j++) oacc[i][j] = z;

    for (int sb = 0; sb < 4096; sb += FA_BS) {
        const int p = (sb >> 6) & 1;
        __bf16* Ksc = p ? Ks1 : Ks0;     // current K block
        __bf16* Ksn = p ? Ks0 : Ks1;     // next K block (TDM target)

#if !USE_TDM
        load_k_block_manual(Ksc, K + boff + (size_t)sb * FA_D, tid);
#endif
        // Load V block transposed: Vt[d][s] = V[sb+s][d].
        for (int i = tid; i < FA_BS * (FA_D / 8); i += 256) {
            int s = i >> 6, c = i & 63;
            bf16x8 v = *(const bf16x8*)(V + boff + (size_t)(sb + s) * FA_D + c * 8);
#pragma unroll
            for (int e = 0; e < 8; e++) Vt[(c * 8 + e) * VT_STR + s] = v[e];
        }
        // Prefetch next V block into cache while we compute this S block.
        if (sb + FA_BS < 4096)
            __builtin_prefetch(V + boff + (size_t)(sb + FA_BS) * FA_D + (size_t)tid * 128, 0, 1);

#if USE_TDM
        if (wave == 0) __builtin_amdgcn_s_wait_tensorcnt(0);  // K[sb] -> Ksc done
#endif
        __syncthreads();
#if USE_TDM
        if (wave == 0 && sb + FA_BS < 4096)   // DMA next K block during compute
            tdm_load_k_tile((unsigned)(uintptr_t)Ksn, K + boff + (size_t)(sb + FA_BS) * FA_D);
#else
        (void)Ksn;
#endif

        // --- S = Q * K^T : 16 tiles of 16x16 over 8 waves (2 tiles each).
        {
            const int t0 = wave * 2;
            f32x8 sacc[2] = {z, z};
            for (int k0 = 0; k0 < FA_D; k0 += 32) {
#pragma unroll
                for (int t = 0; t < 2; t++) {
                    const int mi = (t0 + t) >> 2, ni = (t0 + t) & 3;
                    bf16x16 a  = load_frag_a(Qs,  QS_STR, mi * 16, k0, lane);
                    bf16x16 bb = load_frag_b(Ksc, QS_STR, ni * 16, k0, lane);
                    sacc[t] = wmma_bf16(a, bb, sacc[t]);
                }
            }
            const int khalf = lane >> 4, colL = lane & 15;
#pragma unroll
            for (int t = 0; t < 2; t++) {
                const int mi = (t0 + t) >> 2, ni = (t0 + t) & 3;
#pragma unroll
                for (int r = 0; r < 8; r++)
                    Ss[(mi * 16 + r + 8 * khalf) * SS_STR + ni * 16 + colL] = sacc[t][r];
            }
        }
        __syncthreads();

        // --- Online softmax: 4 threads per row, 16 cols each.
        {
            const int row = tid >> 2, sub = tid & 3;
            float tmax = -1e30f;
            for (int c = sub * 16; c < sub * 16 + 16; c++)
                tmax = fmaxf(tmax, Ss[row * SS_STR + c]);
            red[row * 4 + sub] = tmax;
            __syncthreads();
            const float mo = mrow[row];
            const float mn = fmaxf(mo, fmaxf(fmaxf(red[row * 4 + 0], red[row * 4 + 1]),
                                             fmaxf(red[row * 4 + 2], red[row * 4 + 3])));
            const float alpha = __expf(mo - mn);
            float psum = 0.f;
            for (int c = sub * 16; c < sub * 16 + 16; c++) {
                float pv = __expf(Ss[row * SS_STR + c] - mn);
                Ps[row * PS_STR + c] = f2bf(pv);
                psum += pv;
            }
            __syncthreads();
            red[row * 4 + sub] = psum;
            if (sub == 0) { mrow[row] = mn; arow[row] = alpha; }
            __syncthreads();
            if (sub == 0)
                lrow[row] = lrow[row] * alpha +
                            (red[row * 4 + 0] + red[row * 4 + 1] +
                             red[row * 4 + 2] + red[row * 4 + 3]);
        }
        __syncthreads();

        // --- O = alpha*O + P @ V  (wave's 64-wide d-slice).
        {
            const int khalf = lane >> 4;
#pragma unroll
            for (int mi = 0; mi < 4; mi++)
#pragma unroll
                for (int ni = 0; ni < 4; ni++)
#pragma unroll
                    for (int r = 0; r < 8; r++) {
                        float a = arow[mi * 16 + r + 8 * khalf];
                        oacc[mi][ni][r] = oacc[mi][ni][r] * a;
                    }
#pragma unroll
            for (int k0 = 0; k0 < FA_BS; k0 += 32) {
#pragma unroll
                for (int mi = 0; mi < 4; mi++) {
                    bf16x16 a = load_frag_a(Ps, PS_STR, mi * 16, k0, lane);
#pragma unroll
                    for (int ni = 0; ni < 4; ni++) {
                        bf16x16 bb = load_frag_b(Vt, VT_STR, wave * 64 + ni * 16, k0, lane);
                        oacc[mi][ni] = wmma_bf16(a, bb, oacc[mi][ni]);
                    }
                }
            }
        }
        __syncthreads();   // protect Vt/Ss/Ps (and K buffer rotation) for next block
    }

    // Epilogue: normalize by l, write bf16 context.
    {
        const int khalf = lane >> 4, colL = lane & 15;
#pragma unroll
        for (int mi = 0; mi < 4; mi++) {
#pragma unroll
            for (int r = 0; r < 8; r++) {
                const int row = mi * 16 + r + 8 * khalf;
                const float inv = 1.0f / lrow[row];
#pragma unroll
                for (int ni = 0; ni < 4; ni++) {
                    Ctx[boff + (size_t)(q0 + row) * FA_D + wave * 64 + ni * 16 + colL] =
                        f2bf(oacc[mi][ni][r] * inv);
                }
            }
        }
    }
}

// ===========================================================================
// Host-side orchestration.
// ===========================================================================
extern "C" void kernel_launch(void* const* d_in, const int* in_sizes, int n_in,
                              void* d_out, int out_size, void* d_ws, size_t ws_size,
                              hipStream_t stream) {
    (void)in_sizes; (void)n_in; (void)out_size; (void)ws_size;
    const float* query  = (const float*)d_in[0];
    const float* target = (const float*)d_in[1];  // raw reshape: flat [B*4096, 512]
    const float* wq = (const float*)d_in[2];
    const float* bq = (const float*)d_in[3];
    const float* wk = (const float*)d_in[4];
    const float* bk = (const float*)d_in[5];
    const float* wv = (const float*)d_in[6];
    const float* bv = (const float*)d_in[7];
    const float* wo = (const float*)d_in[8];
    const float* bo = (const float*)d_in[9];
    float* out = (float*)d_out;

    const size_t NTOK = (size_t)4 * 4096 * 512;   // 8,388,608
    __bf16* Qb = (__bf16*)d_ws;
    __bf16* Kb = Qb + NTOK;
    __bf16* Vb = Kb + NTOK;
    __bf16* Cb = Vb + NTOK;

    const dim3 gproj(512 / 128, 16384 / 128);     // (4, 128)

    gemm_bias_wmma<float, __bf16><<<gproj, 256, 0, stream>>>(query,  wq, bq, Qb);
    gemm_bias_wmma<float, __bf16><<<gproj, 256, 0, stream>>>(target, wk, bk, Kb);
    gemm_bias_wmma<float, __bf16><<<gproj, 256, 0, stream>>>(target, wv, bv, Vb);

    flash_attn_wmma<<<dim3(4096 / FA_BQ, 4), 256, FA_SMEM, stream>>>(Qb, Kb, Vb, Cb);

    gemm_bias_wmma<__bf16, float><<<gproj, 256, 0, stream>>>(Cb, wo, bo, out);
}
